// SparseTransformer_37374805410088
// MI455X (gfx1250) — matrix operand order repaired and checked by reading, not compiled, not run on hardware
//
#include <hip/hip_runtime.h>
#include <hip/hip_bf16.h>
#include <stdint.h>

// ---- model dims (fixed by reference) ----
#define LL   2
#define EE   512
#define HH   8
#define DD   64
#define TTOK 4096
#define BBAT 2
#define FFD  2048
#define NWIN 33   // 2*16+1
#define WINR 16

typedef __attribute__((ext_vector_type(16))) __bf16 v16bf;
typedef __attribute__((ext_vector_type(8)))  __bf16 v8bf;
typedef __attribute__((ext_vector_type(8)))  float  v8f;
typedef __attribute__((ext_vector_type(4)))  int    v4i;

#if __has_builtin(__builtin_amdgcn_global_load_async_to_lds_b128)
#define HAVE_ASYNC 1
typedef __attribute__((address_space(1))) v4i* gv4i_p;   // global int4*
typedef __attribute__((address_space(3))) v4i* lv4i_p;   // LDS int4*
#else
#define HAVE_ASYNC 0
#endif

// 16-byte global->LDS stage: async (ASYNCcnt-tracked) if available, else direct copy.
__device__ __forceinline__ void cp16(const __bf16* g, __bf16* l) {
#if HAVE_ASYNC
    __builtin_amdgcn_global_load_async_to_lds_b128(
        (gv4i_p)(void*)g, (lv4i_p)(void*)l, 0, 0);
#else
    *(uint4*)l = *(const uint4*)g;
#endif
}

__device__ __forceinline__ void async_join() {
#if HAVE_ASYNC
#if __has_builtin(__builtin_amdgcn_s_wait_asynccnt)
    __builtin_amdgcn_s_wait_asynccnt(0);
#else
    asm volatile("s_wait_asynccnt 0" ::: "memory");
#endif
#endif
}

// ---------------- f32 -> bf16 transposing weight convert ----------------
// in: (L, Kd, Nd) f32 row-major; out: (L, Nd, Kd) bf16 row-major (i.e. W^T)
__global__ __launch_bounds__(256)
void cvt_t_k(const float* __restrict__ in, __bf16* __restrict__ out, int Kd, int Nd) {
    int i = blockIdx.x * 256 + threadIdx.x;
    int n = i % Nd;
    int k = (i / Nd) % Kd;
    int l = i / (Nd * Kd);
    out[((size_t)l * Nd + n) * Kd + k] = (__bf16)in[i];
}

// ---------------- embedding gather: X = tok_emb[tokens] + pos_emb ----------------
__global__ __launch_bounds__(256)
void embed_k(const int* __restrict__ tok, const float* __restrict__ te,
             const float* __restrict__ pe, float* __restrict__ x) {
    int i  = blockIdx.x * 256 + threadIdx.x;      // over B*T*E
    int e  = i & (EE - 1);
    int bt = i >> 9;                              // E = 512
    int t  = bt & (TTOK - 1);
    x[i] = te[(size_t)tok[bt] * EE + e] + pe[(size_t)t * EE + e];
}

// ---------------- LayerNorm (wave per row), writes bf16 ----------------
__global__ __launch_bounds__(256)
void ln_k(const float* __restrict__ x, const float* __restrict__ g,
          const float* __restrict__ b, __bf16* __restrict__ out) {
    int row  = (blockIdx.x * 256 + threadIdx.x) >> 5;   // B*T rows
    int lane = threadIdx.x & 31;
    const float* xr = x + (size_t)row * EE;
    float v[16];
    float s = 0.f;
#pragma unroll
    for (int i = 0; i < 16; i++) { v[i] = xr[lane + i * 32]; s += v[i]; }
#pragma unroll
    for (int o = 16; o >= 1; o >>= 1) s += __shfl_xor(s, o, 32);
    float mean = s * (1.0f / EE);
    float q = 0.f;
#pragma unroll
    for (int i = 0; i < 16; i++) { float d = v[i] - mean; q += d * d; }
#pragma unroll
    for (int o = 16; o >= 1; o >>= 1) q += __shfl_xor(q, o, 32);
    float rs = rsqrtf(q * (1.0f / EE) + 1e-5f);
    __bf16* orow = out + (size_t)row * EE;
#pragma unroll
    for (int i = 0; i < 16; i++) {
        int e = lane + i * 32;
        orow[e] = (__bf16)((v[i] - mean) * rs * g[e] + b[e]);
    }
}

// ---------------- sliding-window attention: wave per (b,h,t), writes bf16 ----------------
__global__ __launch_bounds__(128)
void attn_k(const float* __restrict__ Q, const float* __restrict__ K,
            const float* __restrict__ V, const int* __restrict__ am,
            __bf16* __restrict__ out) {
    int g    = (blockIdx.x * 128 + threadIdx.x) >> 5;   // B*H*T waves
    int lane = threadIdx.x & 31;
    int t = g & (TTOK - 1);
    int h = (g >> 12) & (HH - 1);
    int b = g >> 15;
    size_t base = ((size_t)(b * TTOK + t) * EE) + h * DD;
    float q0 = Q[base + lane];
    float q1 = Q[base + lane + 32];
    float s[NWIN];
#pragma unroll
    for (int w = 0; w < NWIN; w++) {
        int tk = t + w - WINR;
        bool ok = (tk >= 0) && (tk < TTOK);
        int tkc = ok ? tk : 0;
        ok = ok && (am[b * TTOK + tkc] > 0);
        size_t kb = ((size_t)(b * TTOK + tkc) * EE) + h * DD;
        float p = q0 * K[kb + lane] + q1 * K[kb + lane + 32];
#pragma unroll
        for (int o = 16; o >= 1; o >>= 1) p += __shfl_xor(p, o, 32);
        s[w] = ok ? p * 0.125f : -1e9f;   // 1/sqrt(64)
    }
    float m = -1e30f;
#pragma unroll
    for (int w = 0; w < NWIN; w++) m = fmaxf(m, s[w]);
    float sum = 0.f;
#pragma unroll
    for (int w = 0; w < NWIN; w++) { s[w] = __expf(s[w] - m); sum += s[w]; }
    float inv = 1.0f / sum;
    float o0 = 0.f, o1 = 0.f;
#pragma unroll
    for (int w = 0; w < NWIN; w++) {
        int tk = t + w - WINR;
        int tkc = (tk < 0) ? 0 : ((tk >= TTOK) ? (TTOK - 1) : tk);
        size_t vb = ((size_t)(b * TTOK + tkc) * EE) + h * DD;
        float a = s[w] * inv;            // invalid slots: exp(-1e9-m) == 0
        o0 += a * V[vb + lane];
        o1 += a * V[vb + lane + 32];
    }
    out[base + lane]      = (__bf16)o0;
    out[base + lane + 32] = (__bf16)o1;
}

// ---------------- bf16 WMMA GEMM with transposed weights ----------------
// C[M,N] = A[M,K] @ W^T[N,K]^T  (+bias)(+res)(relu)
// Block: 128 thr = 4 waves. Block tile 128M x 64N x 64K, double-buffered async LDS staging.
// Each wave computes 32M x 64N (8 accumulators) so every B fragment feeds 2 wmmas.
// NMAT=3 fuses three weight matrices (Q,K,V) into one launch; each block stages A for one of them.
// All dims are multiples of tile sizes here (M=8192, N in {512,2048}, K in {512,2048}).
template <int NMAT, bool BIAS, bool RES, bool RELU, bool OUTBF>
__global__ __launch_bounds__(128)
void gemm_k(const __bf16* __restrict__ A,
            const __bf16* B0, const __bf16* B1, const __bf16* B2,
            const float* __restrict__ bias, const float* res,
            float* oF0, float* oF1, float* oF2, __bf16* oB0,
            int M, int N, int K, int nTiles) {
    __shared__ __bf16 sA[2][128 * 72];   // [m][k], padded stride 72
    __shared__ __bf16 sB[2][64 * 72];    // [n][k] (weights pre-transposed), stride 72

    const int tid  = threadIdx.x;
    const int mi   = (NMAT > 1) ? (int)(blockIdx.x / nTiles) : 0;
    const int n0   = ((NMAT > 1) ? (int)(blockIdx.x % nTiles) : (int)blockIdx.x) * 64;
    const int m0   = blockIdx.y * 128;
    const int lane = tid & 31;
    const int r    = lane & 15;
    const int half = lane >> 4;
    const int warp = tid >> 5;           // 0..3, wave owns rows [32*warp, 32*warp+32)

    const __bf16* Bw = (mi == 0) ? B0 : (mi == 1 ? B1 : B2);
    float*        oF = (mi == 0) ? oF0 : (mi == 1 ? oF1 : oF2);

    // stage one 128x64 A tile + 64x64 B tile into buffer `buf` (pure 16B chunks)
    auto stageTiles = [&](int k0, int buf) {
#pragma unroll
        for (int i = 0; i < 8; i++) {
            int c = tid + i * 128;          // 1024 chunks of 8 bf16
            int row = c >> 3, seg = (c & 7) * 8;
            cp16(A + (size_t)(m0 + row) * K + k0 + seg, &sA[buf][row * 72 + seg]);
        }
#pragma unroll
        for (int i = 0; i < 4; i++) {
            int c = tid + i * 128;          // 512 chunks of 8 bf16
            int row = c >> 3, seg = (c & 7) * 8;
            cp16(Bw + (size_t)(n0 + row) * K + k0 + seg, &sB[buf][row * 72 + seg]);
        }
    };

    v8f acc[2][4] = {};                     // [row-group][n-subtile]
    stageTiles(0, 0);
    async_join();
    __syncthreads();

    int buf = 0;
    for (int k0 = 0; k0 < K; k0 += 64) {
        if (k0 + 64 < K) stageTiles(k0 + 64, buf ^ 1);   // prefetch next stage (async)

        union F { v16bf v; v8bf h[2]; };
        // A fragments: 2 row-groups x 2 K-steps.
        // lanes 0-15 hold K 0-7 & 16-23 of row; lanes 16-31 hold K 8-15 & 24-31
        F af[2][2];
#pragma unroll
        for (int ag = 0; ag < 2; ag++) {
            const __bf16* aBase = &sA[buf][(warp * 32 + ag * 16 + r) * 72];
#pragma unroll
            for (int s = 0; s < 2; s++) {
                af[ag][s].h[0] = *(const v8bf*)(aBase + s * 32 + half * 8);
                af[ag][s].h[1] = *(const v8bf*)(aBase + s * 32 + 16 + half * 8);
            }
        }
#pragma unroll
        for (int j = 0; j < 4; j++) {
            // B fragment: lanes 0-15 -> K 0-7 & 16-23 of col (n0+16j+r); lanes 16-31 -> K 8-15 & 24-31
            const __bf16* bBase = &sB[buf][(j * 16 + r) * 72 + half * 8];
#pragma unroll
            for (int s = 0; s < 2; s++) {
                F bf_;
                bf_.h[0] = *(const v8bf*)(bBase + s * 32);
                bf_.h[1] = *(const v8bf*)(bBase + s * 32 + 16);
#pragma unroll
                for (int ag = 0; ag < 2; ag++) {
                    acc[ag][j] = __builtin_amdgcn_wmma_f32_16x16x32_bf16(
                        false, af[ag][s].v, false, bf_.v, (short)0, acc[ag][j], false, false);
                }
            }
        }
        async_join();      // next-stage tiles landed
        __syncthreads();   // everyone done reading buf / writing buf^1
        buf ^= 1;
    }

    // epilogue: VGPR rr -> M = rr (lanes 0-15) / rr+8 (lanes 16-31); N = n0+16j+r
#pragma unroll
    for (int ag = 0; ag < 2; ag++) {
#pragma unroll
        for (int j = 0; j < 4; j++) {
#pragma unroll
            for (int rr = 0; rr < 8; rr++) {
                int row = m0 + warp * 32 + ag * 16 + rr + half * 8;
                int col = n0 + j * 16 + r;
                float v = acc[ag][j][rr];
                if (BIAS) v += bias[col];
                if (RES)  v += res[(size_t)row * N + col];
                if (RELU) v = fmaxf(v, 0.f);
                if (OUTBF) oB0[(size_t)row * N + col] = (__bf16)v;
                else       oF [(size_t)row * N + col] = v;
            }
        }
    }
}

// ---------------- host orchestration ----------------
extern "C" void kernel_launch(void* const* d_in, const int* in_sizes, int n_in,
                              void* d_out, int out_size, void* d_ws, size_t ws_size,
                              hipStream_t stream) {
    const int*   tokens = (const int*)d_in[0];
    const int*   amask  = (const int*)d_in[1];
    const float* tokemb = (const float*)d_in[2];
    const float* posemb = (const float*)d_in[3];
    const float* Wq  = (const float*)d_in[4];
    const float* Wk  = (const float*)d_in[5];
    const float* Wv  = (const float*)d_in[6];
    const float* Wo  = (const float*)d_in[7];
    const float* bo  = (const float*)d_in[8];
    const float* ln1g = (const float*)d_in[9];
    const float* ln1b = (const float*)d_in[10];
    const float* ln2g = (const float*)d_in[11];
    const float* ln2b = (const float*)d_in[12];
    const float* W1  = (const float*)d_in[13];
    const float* b1  = (const float*)d_in[14];
    const float* W2  = (const float*)d_in[15];
    const float* b2  = (const float*)d_in[16];

    const size_t NT = (size_t)BBAT * TTOK;        // 8192 rows
    float* X = (float*)d_out;                     // running residual lives in d_out

    char* ws = (char*)d_ws;
    size_t off = 0;
    auto salloc = [&](size_t bytes) -> void* {
        void* p = ws + off;
        off += (bytes + 255) & ~(size_t)255;
        return p;
    };
    __bf16* HB  = (__bf16*)salloc(NT * EE * 2);        // LN output (bf16)
    float*  Qb  = (float*) salloc(NT * EE * 4);
    float*  Kb  = (float*) salloc(NT * EE * 4);
    float*  Vb  = (float*) salloc(NT * EE * 4);
    __bf16* AOB = (__bf16*)salloc(NT * EE * 2);        // attention out (bf16)
    __bf16* T1B = (__bf16*)Qb;                         // FFN mid (bf16) reuses Q/K/V region
    __bf16* WqT = (__bf16*)salloc((size_t)LL * EE * EE * 2);   // all weights stored transposed [N][K]
    __bf16* WkT = (__bf16*)salloc((size_t)LL * EE * EE * 2);
    __bf16* WvT = (__bf16*)salloc((size_t)LL * EE * EE * 2);
    __bf16* WoT = (__bf16*)salloc((size_t)LL * EE * EE * 2);
    __bf16* W1T = (__bf16*)salloc((size_t)LL * EE * FFD * 2);
    __bf16* W2T = (__bf16*)salloc((size_t)LL * FFD * EE * 2);

    // weight conversion f32 -> bf16 with transpose
    {
        int nEE = LL * EE * EE;       // 524288
        int nEF = LL * EE * FFD;      // 2097152
        cvt_t_k<<<nEE / 256, 256, 0, stream>>>(Wq, WqT, EE, EE);
        cvt_t_k<<<nEE / 256, 256, 0, stream>>>(Wk, WkT, EE, EE);
        cvt_t_k<<<nEE / 256, 256, 0, stream>>>(Wv, WvT, EE, EE);
        cvt_t_k<<<nEE / 256, 256, 0, stream>>>(Wo, WoT, EE, EE);
        cvt_t_k<<<nEF / 256, 256, 0, stream>>>(W1, W1T, EE, FFD);
        cvt_t_k<<<nEF / 256, 256, 0, stream>>>(W2, W2T, FFD, EE);
    }

    // embedding
    embed_k<<<(int)(NT * EE / 256), 256, 0, stream>>>(tokens, tokemb, posemb, X);

    dim3 gQKV(3 * (EE / 64), (int)(NT / 128));  // fused Q,K,V
    dim3 gProj(EE / 64, (int)(NT / 128));       // N=512 GEMMs
    dim3 gFF1 (FFD / 64, (int)(NT / 128));      // N=2048 GEMM
    int lnBlocks   = (int)(NT * 32 / 256);
    int attnBlocks = (int)(NT * HH / 4);        // wave per (b,h,t), 4 waves/block

    for (int l = 0; l < LL; l++) {
        const __bf16* wq = WqT + (size_t)l * EE * EE;
        const __bf16* wk = WkT + (size_t)l * EE * EE;
        const __bf16* wv = WvT + (size_t)l * EE * EE;
        const __bf16* wo = WoT + (size_t)l * EE * EE;
        const __bf16* w1 = W1T + (size_t)l * EE * FFD;
        const __bf16* w2 = W2T + (size_t)l * FFD * EE;

        // h = LN1(x)
        ln_k<<<lnBlocks, 256, 0, stream>>>(X, ln1g + l * EE, ln1b + l * EE, HB);
        // q,k,v projections (fused: one launch; each block uses one of the 3 weight matrices)
        gemm_k<3, false, false, false, false><<<gQKV, 128, 0, stream>>>(
            HB, wq, wk, wv, nullptr, nullptr, Qb, Kb, Vb, nullptr, (int)NT, EE, EE, EE / 64);
        // sliding-window attention
        attn_k<<<attnBlocks, 128, 0, stream>>>(Qb, Kb, Vb, amask, AOB);
        // x = x + attn @ Wo + bo
        gemm_k<1, true, true, false, false><<<gProj, 128, 0, stream>>>(
            AOB, wo, wo, wo, bo + l * EE, X, X, nullptr, nullptr, nullptr, (int)NT, EE, EE, EE / 64);
        // h = LN2(x)
        ln_k<<<lnBlocks, 256, 0, stream>>>(X, ln2g + l * EE, ln2b + l * EE, HB);
        // t1 = relu(h @ W1 + b1)  (bf16)
        gemm_k<1, true, false, true, true><<<gFF1, 128, 0, stream>>>(
            HB, w1, w1, w1, b1 + l * FFD, nullptr, nullptr, nullptr, nullptr, T1B, (int)NT, FFD, EE, FFD / 64);
        // x = x + t1 @ W2 + b2
        gemm_k<1, true, true, false, false><<<gProj, 128, 0, stream>>>(
            T1B, w2, w2, w2, b2 + l * EE, X, X, nullptr, nullptr, nullptr, (int)NT, EE, FFD, EE / 64);
    }
    (void)in_sizes; (void)n_in; (void)out_size; (void)ws_size;
}
